// utm_8847632629824
// MI455X (gfx1250) — matrix operand order repaired and checked by reading, not compiled, not run on hardware
//
#include <hip/hip_runtime.h>
#include <hip/hip_bf16.h>
#include <math.h>

typedef __attribute__((ext_vector_type(16))) _Float16 v16h;
typedef __attribute__((ext_vector_type(8)))  _Float16 v8h;
typedef __attribute__((ext_vector_type(8)))  float    v8f;

#define HW   16384      // 128*128
#define C0   256
#define C1   128
#define C2   32
#define NB   8
#define NT   64         // pixels per tile (k_net / k_final)
#define XSTR 272        // C0 + 16 pad  (544 B row = 17*32)
#define HSTR 144        // C1 + 16 pad  (288 B row = 9*32)
#define WSTR 48         // weff LDS stride (96 B row = 3*32)
#define CSTR 48         // k_final Cs stride (96 B row = 3*32)

// ---------------------------------------------------------------------------
// K0: per-(b,c) stats: content mean + unbiased rstd, style mean
// ---------------------------------------------------------------------------
__global__ __launch_bounds__(256) void k_stats(
    const float* __restrict__ content, const float* __restrict__ style,
    float* __restrict__ cmean, float* __restrict__ crstd, float* __restrict__ smean)
{
    const int bc = blockIdx.x;                       // 0..2047
    const float* cp = content + (size_t)bc * HW;
    const float* sp = style   + (size_t)bc * HW;
    float cs = 0.f, css = 0.f, ss = 0.f;
    for (int i = threadIdx.x; i < HW; i += 256) {
        float cv = cp[i], sv = sp[i];
        cs += cv; css += cv * cv; ss += sv;
    }
    __shared__ float r0[256], r1[256], r2[256];
    r0[threadIdx.x] = cs; r1[threadIdx.x] = css; r2[threadIdx.x] = ss;
    __syncthreads();
    for (int s = 128; s > 0; s >>= 1) {
        if (threadIdx.x < s) {
            r0[threadIdx.x] += r0[threadIdx.x + s];
            r1[threadIdx.x] += r1[threadIdx.x + s];
            r2[threadIdx.x] += r2[threadIdx.x + s];
        }
        __syncthreads();
    }
    if (threadIdx.x == 0) {
        float mean = r0[0] * (1.f / HW);
        float var  = (r1[0] - r0[0] * r0[0] * (1.f / HW)) * (1.f / (HW - 1));
        cmean[bc] = mean;
        crstd[bc] = rsqrtf(var + 1e-5f);
        smean[bc] = r2[0] * (1.f / HW);
    }
}

// ---------------------------------------------------------------------------
// K1: fused net(): normalize -> conv1x1(256->128)+ReLU -> conv1x1(128->32)
//     for content (inp=0) and style (inp=1).
//     Output F16[2][8][HW][32] (pixel-major f16, 16B contiguous per store).
// ---------------------------------------------------------------------------
__global__ __launch_bounds__(256) void k_net(
    const float* __restrict__ content, const float* __restrict__ style,
    const float* __restrict__ w1, const float* __restrict__ b1,
    const float* __restrict__ w2, const float* __restrict__ b2,
    const float* __restrict__ cmean, const float* __restrict__ crstd,
    const float* __restrict__ smean, _Float16* __restrict__ F16)
{
    __shared__ __align__(64) _Float16 Xs[NT * XSTR];   // pixel-major normalized input
    __shared__ __align__(64) _Float16 Hs[NT * HSTR];   // pixel-major hidden

    const int blk  = blockIdx.x;
    const int tile = blk & 255;            // 256 tiles of 64 px
    const int b    = (blk >> 8) & 7;
    const int inp  = blk >> 11;            // 0=content, 1=style
    const int px0  = tile * NT;
    const int tid  = threadIdx.x;

    const float* src = (inp == 0) ? content : style;
    const size_t base = (size_t)b * C0 * HW;

    // ---- stage + normalize + cvt f16 into LDS (pixel-major) ----
    #pragma unroll 4
    for (int i = 0; i < (C0 * NT) / 256; ++i) {        // 64 iters
        int idx = i * 256 + tid;
        int c = idx >> 6;                              // /NT
        int p = idx & (NT - 1);
        float v = src[base + (size_t)c * HW + px0 + p];
        int bc = b * C0 + c;
        float x = (inp == 0) ? (v - cmean[bc]) * crstd[bc] : (v - smean[bc]);
        Xs[p * XSTR + c] = (_Float16)x;
    }
    __syncthreads();

    const int wave = tid >> 5;
    const int lane = tid & 31;
    const int half = lane >> 4;
    const int l15  = lane & 15;

    // ---- layer 1: H[128,NT] = relu(W1 @ X + b1), wave = M-tile ----
    {
        const int mt = wave;                           // 0..7
        v16h afrag[8];
        #pragma unroll
        for (int kt = 0; kt < 8; ++kt) {               // K = 256 in 8 steps of 32
            const float* wr = w1 + (size_t)(mt * 16 + l15) * C0 + kt * 32 + half * 16;
            v16h a;
            #pragma unroll
            for (int e = 0; e < 8; ++e) a[e]     = (_Float16)wr[e];
            #pragma unroll
            for (int e = 0; e < 8; ++e) a[8 + e] = (_Float16)wr[8 + e];
            afrag[kt] = a;
        }
        #pragma unroll
        for (int nt = 0; nt < 4; ++nt) {
            v8f acc = {};
            const int n = nt * 16 + l15;
            #pragma unroll
            for (int kt = 0; kt < 8; ++kt) {
                const v16h bf = *(const v16h*)&Xs[n * XSTR + kt * 32 + half * 16];
                acc = __builtin_amdgcn_wmma_f32_16x16x32_f16(
                        false, afrag[kt], false, bf, (short)0, acc, false, false);
            }
            const int mb = mt * 16 + half * 8;
            v8h hv;
            #pragma unroll
            for (int v = 0; v < 8; ++v) {
                float y = acc[v] + b1[mb + v];
                hv[v] = (_Float16)(y > 0.f ? y : 0.f);
            }
            *(v8h*)&Hs[n * HSTR + mb] = hv;            // 16B aligned store
        }
    }
    __syncthreads();

    // ---- layer 2: Y[32,NT] = W2 @ H + b2, 8 waves -> 2x4 tiles ----
    {
        const int mt = wave >> 2;                      // 0..1
        const int nt = wave & 3;                       // 0..3
        const int n  = nt * 16 + l15;
        v8f acc = {};
        #pragma unroll
        for (int kt = 0; kt < 4; ++kt) {               // K = 128
            const float* wr = w2 + (size_t)(mt * 16 + l15) * C1 + kt * 32 + half * 16;
            v16h a;
            #pragma unroll
            for (int e = 0; e < 8; ++e) a[e]     = (_Float16)wr[e];
            #pragma unroll
            for (int e = 0; e < 8; ++e) a[8 + e] = (_Float16)wr[8 + e];
            const v16h bf = *(const v16h*)&Hs[n * HSTR + kt * 32 + half * 16];
            acc = __builtin_amdgcn_wmma_f32_16x16x32_f16(
                    false, a, false, bf, (short)0, acc, false, false);
        }
        const int mb = mt * 16 + half * 8;
        v8h yv;
        #pragma unroll
        for (int v = 0; v < 8; ++v)
            yv[v] = (_Float16)(acc[v] + b2[mb + v]);
        // pixel-major f16: [inp][b][px][32] -> one contiguous 16B store per lane
        _Float16* dst = F16 + (((size_t)inp * NB + b) * HW + px0 + n) * C2 + mb;
        *(v8h*)dst = yv;
    }
}

// ---------------------------------------------------------------------------
// K2: s_cov = softmax(sF sF^T / hw) * 32^-0.5 ; weff = wu @ s_cov
//     sF16 is pixel-major f16 [HW][32] per batch -> fully contiguous chunks.
// ---------------------------------------------------------------------------
__global__ __launch_bounds__(256) void k_cov(
    const _Float16* __restrict__ sF16,  // [8][HW][32]
    const float* __restrict__ wu,       // [256][32]
    float* __restrict__ weff)           // [8][256][32]
{
    __shared__ _Float16 Sh[256 * 32];   // 16KB chunk: 256 px x 32 ch, pixel-major
    __shared__ float cov[32][33];
    __shared__ float covs[32][33];

    const int b   = blockIdx.x;
    const int tid = threadIdx.x;
    const _Float16* sp = sF16 + (size_t)b * HW * C2;

    float acc[4] = {0.f, 0.f, 0.f, 0.f};
    for (int c0 = 0; c0 < HW; c0 += 256) {
        #pragma unroll 8
        for (int i = 0; i < 32; ++i) {
            int idx = i * 256 + tid;                   // contiguous 16KB
            Sh[idx] = sp[(size_t)c0 * C2 + idx];
        }
        __syncthreads();
        #pragma unroll
        for (int q = 0; q < 4; ++q) {
            int p = tid + 256 * q;                     // pair id 0..1023
            int ii = p >> 5, jj = p & 31;              // ii uniform per wave
            float a = 0.f;
            #pragma unroll 8
            for (int x = 0; x < 256; ++x)
                a += (float)Sh[x * 32 + ii] * (float)Sh[x * 32 + jj];
            acc[q] += a;
        }
        __syncthreads();
    }
    #pragma unroll
    for (int q = 0; q < 4; ++q) {
        int p = tid + 256 * q;
        cov[p >> 5][p & 31] = acc[q] * (1.f / HW);
    }
    __syncthreads();
    if (tid < 32) {                                    // one lane per row: softmax
        float mx = -1e30f;
        #pragma unroll
        for (int j = 0; j < 32; ++j) mx = fmaxf(mx, cov[tid][j]);
        float e[32], s = 0.f;
        #pragma unroll
        for (int j = 0; j < 32; ++j) { e[j] = __expf(cov[tid][j] - mx); s += e[j]; }
        float inv = (1.f / s) * 0.17677669529663687f;  // 32^-0.5
        #pragma unroll
        for (int j = 0; j < 32; ++j) covs[tid][j] = e[j] * inv;
    }
    __syncthreads();
    {                                                  // weff row per thread
        const int o = tid;
        float wr[32];
        #pragma unroll
        for (int d = 0; d < 32; ++d) wr[d] = wu[o * 32 + d];
        float* dst = weff + ((size_t)b * 256 + o) * 32;
        #pragma unroll 4
        for (int k = 0; k < 32; ++k) {
            float a = 0.f;
            #pragma unroll
            for (int d = 0; d < 32; ++d) a += wr[d] * covs[d][k];
            dst[k] = a;
        }
    }
}

// ---------------------------------------------------------------------------
// K3: out = weff @ cF + bu + smean + content   (single K=32 WMMA per tile)
//     cF tile staged via CDNA5 async-to-LDS DMA (per-lane LDS addresses add
//     the 96B row padding during the copy; ASYNCcnt tracked).
// ---------------------------------------------------------------------------
__global__ __launch_bounds__(256) void k_final(
    const _Float16* __restrict__ cF16,  // [8][HW][32] pixel-major f16
    const float* __restrict__ weff,     // [8][256][32]
    const float* __restrict__ bu,       // [256]
    const float* __restrict__ smean,    // [8][256]
    const float* __restrict__ content,
    float* __restrict__ out)
{
    __shared__ __align__(64) _Float16 Wf[C0 * WSTR];   // 256x32 f16 (padded)
    __shared__ __align__(64) _Float16 Cs[NT * CSTR];   // 64px x 32ch f16 (padded)
    __shared__ float sb[C0];

    const int blk  = blockIdx.x;
    const int tile = blk & 255;
    const int b    = blk >> 8;
    const int px0  = tile * NT;
    const int tid  = threadIdx.x;

    // ---- async DMA: contiguous 4KB cF tile -> padded LDS rows ----
    {
        const _Float16* gsrc = cF16 + ((size_t)b * HW + px0) * C2;
        const int px   = tid >> 2;                     // 4x 16B units per pixel row
        const int unit = tid & 3;
        unsigned ldst = (unsigned)(size_t)(&Cs[0]) + (unsigned)(px * (CSTR * 2) + unit * 16);
        unsigned long long ga = (unsigned long long)(size_t)gsrc + (unsigned long long)tid * 16;
        asm volatile("global_load_async_to_lds_b128 %0, %1, off"
                     :: "v"(ldst), "v"(ga) : "memory");
    }

    // ---- regular staging of weff (f32->f16) and bias+smean while DMA flies ----
    const float* wsrc = weff + (size_t)b * C0 * C2;
    #pragma unroll 4
    for (int i = 0; i < 32; ++i) {
        int idx = i * 256 + tid;
        Wf[(idx >> 5) * WSTR + (idx & 31)] = (_Float16)wsrc[idx];
    }
    sb[tid] = bu[tid] + smean[b * C0 + tid];

    asm volatile("s_wait_asynccnt 0x0" ::: "memory");
    __syncthreads();

    const int wave = tid >> 5, lane = tid & 31;
    const int half = lane >> 4, l15 = lane & 15;
    const float* cbase = content + (size_t)b * C0 * HW;
    float*       obase = out     + (size_t)b * C0 * HW;

    #pragma unroll
    for (int t = 0; t < 8; ++t) {                      // 64 tiles / 8 waves
        const int ti = wave * 8 + t;
        const int mt = ti >> 2, nt = ti & 3;
        v16h a;
        {
            const _Float16* ar = &Wf[(mt * 16 + l15) * WSTR];
            v8h lo = *(const v8h*)(ar + half * 16);
            v8h hi = *(const v8h*)(ar + half * 16 + 8);
            #pragma unroll
            for (int e = 0; e < 8; ++e) { a[e] = lo[e]; a[8 + e] = hi[e]; }
        }
        const int n = nt * 16 + l15;
        const v16h bf = *(const v16h*)&Cs[n * CSTR + half * 16];
        v8f acc = {};
        acc = __builtin_amdgcn_wmma_f32_16x16x32_f16(
                false, a, false, bf, (short)0, acc, false, false);
        const int mb = mt * 16 + half * 8;
        #pragma unroll
        for (int v = 0; v < 8; ++v) {
            size_t gi = (size_t)(mb + v) * HW + px0 + n;
            obase[gi] = acc[v] + sb[mb + v] + cbase[gi];
        }
    }
}

// ---------------------------------------------------------------------------
extern "C" void kernel_launch(void* const* d_in, const int* in_sizes, int n_in,
                              void* d_out, int out_size, void* d_ws, size_t ws_size,
                              hipStream_t stream)
{
    const float* content = (const float*)d_in[0];
    const float* style   = (const float*)d_in[1];
    const float* w1      = (const float*)d_in[2];
    const float* b1      = (const float*)d_in[3];
    const float* w2      = (const float*)d_in[4];
    const float* b2      = (const float*)d_in[5];
    const float* wu      = (const float*)d_in[6];
    const float* bu      = (const float*)d_in[7];
    float* out = (float*)d_out;

    float* ws    = (float*)d_ws;
    float* cmean = ws;                                  // 2048 f32
    float* crstd = ws + 2048;                           // 2048 f32
    float* smean = ws + 4096;                           // 2048 f32
    _Float16* F16 = (_Float16*)(ws + 6144);             // [2][8][HW][32] f16 = 16MB
    const size_t F_ELEMS = (size_t)2 * NB * HW * C2;    // 8388608 f16
    float* weff  = ws + 6144 + (F_ELEMS / 2);           // [8][256][32] f32
    _Float16* cF16 = F16;
    _Float16* sF16 = F16 + (size_t)NB * HW * C2;

    k_stats<<<NB * C0, 256, 0, stream>>>(content, style, cmean, crstd, smean);
    k_net  <<<2 * NB * (HW / NT), 256, 0, stream>>>(content, style, w1, b1, w2, b2,
                                                    cmean, crstd, smean, F16);
    k_cov  <<<NB, 256, 0, stream>>>(sF16, wu, weff);
    k_final<<<NB * (HW / NT), 256, 0, stream>>>(cF16, weff, bu, smean, content, out);
}
